// Mamba2LM_738734375260
// MI455X (gfx1250) — compile-verified, matrix operand order checked by reading, not run on hardware
//
#include <hip/hip_runtime.h>

// ---- model dims ----
#define VOCAB   32000
#define DIMM    512
#define DEPTHL  2
#define NSTATE  64
#define PHEAD   64
#define DINN    1024
#define NHEAD   16
#define DCONVK  4
#define CONVD   1152
#define DPROJJ  2192
#define DPROJP  2304                 // DPROJ padded to 256-col block tile
#define BATCHB  2
#define SEQS    1024
#define ROWSM   (BATCHB * SEQS)      // 2048
#define EPSF    1e-5f

typedef __bf16 bf16;
typedef __attribute__((ext_vector_type(16))) __bf16 v16bf;
typedef __attribute__((ext_vector_type(8)))  float  v8f;
typedef __attribute__((ext_vector_type(4)))  unsigned int u32x4;
typedef __attribute__((ext_vector_type(8)))  int  i32x8;
typedef __attribute__((ext_vector_type(4)))  int  i32x4;

// ---------------- helpers ----------------
__device__ __forceinline__ bf16 f32_to_bf16(float f) {
    union { float f; unsigned u; } a; a.f = f;
    unsigned r = a.u + 0x7FFFu + ((a.u >> 16) & 1u);   // round-to-nearest-even
    union { unsigned short s; bf16 b; } o;
    o.s = (unsigned short)(r >> 16);
    return o.b;
}

__device__ __forceinline__ float siluf(float x) {
    return x / (1.f + __expf(-x));
}

// 16-bit 16x32 fragment per ISA layout, read from an LDS row of 32 bf16:
// lanes 0-15 take K=[0..7],[16..23]; lanes 16-31 take K=[8..15],[24..31].
__device__ __forceinline__ v16bf lds_frag(const bf16* q0, int lane) {
    const bf16* q = q0 + ((lane & 16) ? 8 : 0);
    union { v16bf v; uint4 u[2]; } r;
    r.u[0] = *(const uint4*)(q);
    r.u[1] = *(const uint4*)(q + 16);
    return r.v;
}

// ---- Tensor Data Mover: 2D tile (rows x cols, element bf16) global -> LDS ----
// D# group0: count=1, lds_addr, global_addr, type=2.  group1: data_size=2B,
// tensor dims, tile dims, dim0 stride (elements).  groups 2/3 unused (2D).
__device__ __forceinline__ void tdm_load_2d(const bf16* gsrc, unsigned lds_byte_addr,
                                            int rows, int cols, int pitch_elems) {
    unsigned long long ga = (unsigned long long)(size_t)gsrc;
    u32x4 g0;
    g0[0] = 1u;                                             // count=1, user D#
    g0[1] = lds_byte_addr;                                  // LDS dest (bytes)
    g0[2] = (unsigned)(ga & 0xFFFFFFFFu);                   // global_addr[31:0]
    g0[3] = (unsigned)((ga >> 32) & 0x01FFFFFFu) | (2u << 30); // [56:32] | type=2
    unsigned td0 = (unsigned)pitch_elems;                   // tensor_dim0 (K extent)
    unsigned td1 = (unsigned)rows;                          // tensor_dim1
    i32x8 g1;
    g1[0] = (int)(1u << 16);                                // data_size=1 (2 bytes)
    g1[1] = (int)((td0 & 0xFFFFu) << 16);                   // tensor_dim0 lo
    g1[2] = (int)((td0 >> 16) | ((td1 & 0xFFFFu) << 16));   // td0 hi | td1 lo
    g1[3] = (int)((td1 >> 16) | ((unsigned)cols << 16));    // td1 hi | tile_dim0
    g1[4] = rows & 0xFFFF;                                  // tile_dim1 (tile_dim2=0)
    g1[5] = (int)(unsigned)pitch_elems;                     // tensor_dim0_stride lo
    g1[6] = 0;                                              // stride0 hi | stride1 lo
    g1[7] = 0;
    i32x4 z4 = {0, 0, 0, 0};
    i32x8 z8 = {0, 0, 0, 0, 0, 0, 0, 0};
    __builtin_amdgcn_tensor_load_to_lds(g0, g1, z4, z4, z8, 0);
}

__device__ __forceinline__ unsigned lds_addr_of(const void* p) {
    return (unsigned)(unsigned long long)(size_t)p;         // LDS offset = addr[31:0]
}

// ---------------- fp32 -> bf16 convert ----------------
__global__ void __launch_bounds__(256)
cvt_f32_bf16(const float* __restrict__ src, bf16* __restrict__ dst, size_t n) {
    size_t i = (size_t)blockIdx.x * blockDim.x + threadIdx.x;
    if (i < n) dst[i] = f32_to_bf16(src[i]);
}

// convert w_in [DEPTH, DPROJ, DIM] -> padded [DEPTH, DPROJP, DIM], zero pad rows
__global__ void __launch_bounds__(256)
cvt_w_in_pad(const float* __restrict__ src, bf16* __restrict__ dst) {
    size_t i = (size_t)blockIdx.x * blockDim.x + threadIdx.x;
    if (i >= (size_t)DEPTHL * DPROJP * DIMM) return;
    int k = (int)(i % DIMM);
    int r = (int)((i / DIMM) % DPROJP);
    int l = (int)(i / ((size_t)DIMM * DPROJP));
    float v = (r < DPROJJ) ? src[((size_t)l * DPROJJ + r) * DIMM + k] : 0.f;
    dst[i] = f32_to_bf16(v);
}

// ---------------- embedding gather ----------------
__global__ void __launch_bounds__(256)
embed_gather(const int* __restrict__ tokens, const float* __restrict__ embed,
             float* __restrict__ h, bf16* __restrict__ hbf) {
    int row = blockIdx.x;                         // 0..2047
    int tok = tokens[row];
    const float* src = embed + (size_t)tok * DIMM;
    for (int d = threadIdx.x; d < DIMM; d += blockDim.x) {
        float v = src[d];
        h  [(size_t)row * DIMM + d] = v;
        hbf[(size_t)row * DIMM + d] = f32_to_bf16(v);
    }
}

// ------------- TDM-staged bf16 WMMA GEMM: C[m,n] = sum_k A[m,k]*W[n,k] -------
// A: [M,K] bf16, W: [>=N rows, K] bf16 (rows padded up to block tile), C: [M,N] f32.
// Block: 256 thr = 8 waves as 2x4; block tile 32x256; K-step 32; LDS double buffer
// filled by TENSOR_LOAD_TO_LDS, synced with s_wait_tensorcnt + barrier.
#define BM 32
#define BN 256
#define BK 32
__global__ void __launch_bounds__(256)
gemm_bf16_wmma(const bf16* __restrict__ A, const bf16* __restrict__ W,
               float* __restrict__ C, bf16* Cbf, const float* addsrc,
               int M, int N, int K) {
    __shared__ alignas(16) bf16 Apan[2][BM * BK];   //  2 KB / buf
    __shared__ alignas(16) bf16 Wpan[2][BN * BK];   // 16 KB / buf

    const int lane = threadIdx.x & 31;
    const int wave = threadIdx.x >> 5;     // 0..7
    const int wr   = wave >> 2;            // 0..1: 16-row sub-tile
    const int wc   = wave & 3;             // 0..3: 64-col group
    const int nTiles = N >> 4;

    const bf16* Ablk = A + (size_t)(blockIdx.x * BM) * K;
    const bf16* Wblk = W + (size_t)(blockIdx.y * BN) * K;

    // prologue: DMA first K-panel into buffer 0
    if (threadIdx.x < 32) {
        tdm_load_2d(Ablk, lds_addr_of(&Apan[0][0]), BM, BK, K);
        tdm_load_2d(Wblk, lds_addr_of(&Wpan[0][0]), BN, BK, K);
        __builtin_amdgcn_s_wait_tensorcnt(0);
    }
    __syncthreads();

    v8f acc[4] = {};
    int cur = 0;
    for (int k0 = 0; k0 < K; k0 += BK) {
        int nxt = cur ^ 1;
        // overlap: wave 0 kicks off the next panel's DMA while everyone computes
        if (k0 + BK < K && threadIdx.x < 32) {
            tdm_load_2d(Ablk + k0 + BK, lds_addr_of(&Apan[nxt][0]), BM, BK, K);
            tdm_load_2d(Wblk + k0 + BK, lds_addr_of(&Wpan[nxt][0]), BN, BK, K);
        }
        // compute from current buffer (ds reads, A-fragment reused 4x)
        v16bf af = lds_frag(&Apan[cur][(wr * 16 + (lane & 15)) * BK], lane);
#pragma unroll
        for (int t = 0; t < 4; ++t) {
            v16bf wf = lds_frag(&Wpan[cur][(wc * 64 + t * 16 + (lane & 15)) * BK], lane);
            acc[t] = __builtin_amdgcn_wmma_f32_16x16x32_bf16(
                false, af, false, wf, (short)0, acc[t], false, false);
        }
        if (k0 + BK < K && threadIdx.x < 32)
            __builtin_amdgcn_s_wait_tensorcnt(0);
        __syncthreads();
        cur = nxt;
    }

    // C/D layout: VGPR r holds (M = r + 8*(lane>=16), N = lane%16)
    const int mhi = (lane >> 4) << 3;
    const int nc  = lane & 15;
#pragma unroll
    for (int t = 0; t < 4; ++t) {
        int nt = blockIdx.y * (BN / 16) + wc * 4 + t;
        if (nt >= nTiles) continue;                   // wave-uniform guard
        int n = (nt << 4) + nc;
#pragma unroll
        for (int r = 0; r < 8; ++r) {
            int m = blockIdx.x * BM + wr * 16 + mhi + r;
            float v = acc[t][r];
            size_t idx = (size_t)m * N + n;
            if (addsrc) v += addsrc[idx];
            C[idx] = v;
            if (Cbf) Cbf[idx] = f32_to_bf16(v);
        }
    }
}

// ---------------- causal depthwise conv1d + SiLU over xBC channels --------
__global__ void __launch_bounds__(256)
conv_silu(const float* __restrict__ zxbcdt, const float* __restrict__ cw,
          const float* __restrict__ cb, float* __restrict__ xbc) {
    int idx = blockIdx.x * blockDim.x + threadIdx.x;
    if (idx >= ROWSM * CONVD) return;
    int c   = idx % CONVD;
    int row = idx / CONVD;
    int s   = row % SEQS;
    float acc = cb[c];
#pragma unroll
    for (int k = 0; k < DCONVK; ++k) {
        int sp = s - (DCONVK - 1) + k;
        if (sp >= 0)
            acc += zxbcdt[(size_t)(row - s + sp) * DPROJJ + DINN + c] * cw[c * DCONVK + k];
    }
    xbc[idx] = siluf(acc);
}

// ---------------- sequential SSM scan (one block per (batch, head)) -------
__global__ void __launch_bounds__(256)
ssm_scan(const float* __restrict__ zxbcdt, const float* __restrict__ xbc,
         const float* __restrict__ dt_bias, const float* __restrict__ A_log,
         const float* __restrict__ D_skip, float* __restrict__ yout) {
    const int bh = blockIdx.x;                 // 0..31
    const int b  = bh >> 4, h = bh & 15;
    const int t  = threadIdx.x;
    const int p  = t >> 2;
    const int nq = (t & 3) << 4;

    float state[16];
#pragma unroll
    for (int i = 0; i < 16; ++i) state[i] = 0.f;

    const float Ah  = -__expf(A_log[h]);
    const float dtb = dt_bias[h];
    const float Dh  = D_skip[h];

    __shared__ float Bs[NSTATE], Cs[NSTATE];

    for (int s = 0; s < SEQS; ++s) {
        size_t row = (size_t)b * SEQS + s;
        const float* xr = xbc + row * CONVD;
        if (t < NSTATE)            Bs[t]          = xr[DINN + t];
        else if (t < 2 * NSTATE)   Cs[t - NSTATE] = xr[DINN + NSTATE + (t - NSTATE)];
        __syncthreads();

        float dtraw = zxbcdt[row * DPROJJ + (DINN + CONVD) + h] + dtb;
        float dtv   = (dtraw > 20.f) ? dtraw : log1pf(__expf(dtraw));
        float dA    = __expf(dtv * Ah);
        float xp    = xr[h * PHEAD + p];
        float coef  = dtv * xp;

        float y = 0.f;
#pragma unroll
        for (int i = 0; i < 16; ++i) {
            float st = state[i] * dA + coef * Bs[nq + i];
            state[i] = st;
            y += st * Cs[nq + i];
        }
        y += __shfl_xor(y, 1, 32);
        y += __shfl_xor(y, 2, 32);
        if ((t & 3) == 0)
            yout[row * DINN + h * PHEAD + p] = y + Dh * xp;
        __syncthreads();
    }
}

// ---------------- gated RMSNorm: out = rmsnorm(y * silu(z)) * w -> bf16 ---
__global__ void __launch_bounds__(256)
gated_rmsnorm(const float* __restrict__ zxbcdt, const float* __restrict__ y,
              const float* __restrict__ nw, bf16* __restrict__ out) {
    int row = blockIdx.x;
    __shared__ float red[8];
    float vals[4], ss = 0.f;
#pragma unroll
    for (int i = 0; i < 4; ++i) {
        int d = threadIdx.x + i * 256;
        float z = zxbcdt[(size_t)row * DPROJJ + d];
        float v = y[(size_t)row * DINN + d] * siluf(z);
        vals[i] = v;
        ss += v * v;
    }
    for (int off = 16; off > 0; off >>= 1) ss += __shfl_xor(ss, off, 32);
    int lane = threadIdx.x & 31, w = threadIdx.x >> 5;
    if (lane == 0) red[w] = ss;
    __syncthreads();
    if (threadIdx.x == 0) {
        float tot = 0.f;
        for (int i = 0; i < 8; ++i) tot += red[i];
        red[0] = rsqrtf(tot / DINN + EPSF);
    }
    __syncthreads();
    float sc = red[0];
#pragma unroll
    for (int i = 0; i < 4; ++i) {
        int d = threadIdx.x + i * 256;
        out[(size_t)row * DINN + d] = f32_to_bf16(vals[i] * sc * nw[d]);
    }
}

// ---------------- final RMSNorm (DIM=512) -> bf16 --------------------------
__global__ void __launch_bounds__(256)
final_rmsnorm(const float* __restrict__ h, const float* __restrict__ nf,
              bf16* __restrict__ out) {
    int row = blockIdx.x;
    __shared__ float red[8];
    float vals[2], ss = 0.f;
#pragma unroll
    for (int i = 0; i < 2; ++i) {
        int d = threadIdx.x + i * 256;
        float v = h[(size_t)row * DIMM + d];
        vals[i] = v;
        ss += v * v;
    }
    for (int off = 16; off > 0; off >>= 1) ss += __shfl_xor(ss, off, 32);
    int lane = threadIdx.x & 31, w = threadIdx.x >> 5;
    if (lane == 0) red[w] = ss;
    __syncthreads();
    if (threadIdx.x == 0) {
        float tot = 0.f;
        for (int i = 0; i < 8; ++i) tot += red[i];
        red[0] = rsqrtf(tot / DIMM + EPSF);
    }
    __syncthreads();
    float sc = red[0];
#pragma unroll
    for (int i = 0; i < 2; ++i) {
        int d = threadIdx.x + i * 256;
        out[(size_t)row * DIMM + d] = f32_to_bf16(vals[i] * sc * nf[d]);
    }
}

// ---------------- host orchestration ----------------
extern "C" void kernel_launch(void* const* d_in, const int* in_sizes, int n_in,
                              void* d_out, int out_size, void* d_ws, size_t ws_size,
                              hipStream_t stream) {
    const int*   tokens  = (const int*)  d_in[0];
    const float* embed   = (const float*)d_in[1];
    const float* w_in    = (const float*)d_in[2];
    const float* conv_w  = (const float*)d_in[3];
    const float* conv_b  = (const float*)d_in[4];
    const float* dt_bias = (const float*)d_in[5];
    const float* A_log   = (const float*)d_in[6];
    const float* D_skip  = (const float*)d_in[7];
    const float* norm_w  = (const float*)d_in[8];
    const float* w_out   = (const float*)d_in[9];
    const float* norm_f  = (const float*)d_in[10];
    float* logits = (float*)d_out;

    // workspace layout (256B aligned)
    char* ws = (char*)d_ws;
    size_t off = 0;
    auto alloc = [&](size_t bytes) { char* p = ws + off; off = (off + bytes + 255) & ~(size_t)255; return p; };
    bf16*  embed_bf = (bf16*) alloc((size_t)VOCAB * DIMM * 2);
    bf16*  w_in_bf  = (bf16*) alloc((size_t)DEPTHL * DPROJP * DIMM * 2);  // padded rows
    bf16*  w_out_bf = (bf16*) alloc((size_t)DEPTHL * DIMM * DINN * 2);
    float* h_f      = (float*)alloc((size_t)ROWSM * DIMM * 4);
    bf16*  h_bf     = (bf16*) alloc((size_t)ROWSM * DIMM * 2);
    float* zxbcdt   = (float*)alloc((size_t)ROWSM * DPROJJ * 4);
    float* xbc      = (float*)alloc((size_t)ROWSM * CONVD * 4);
    float* ybuf     = (float*)alloc((size_t)ROWSM * DINN * 4);
    bf16*  yg_bf    = (bf16*) alloc((size_t)ROWSM * DINN * 2);
    bf16*  hn_bf    = (bf16*) alloc((size_t)ROWSM * DIMM * 2);
    (void)ws_size; (void)n_in; (void)in_sizes; (void)out_size;

    // weight conversions
    {
        size_t n = (size_t)VOCAB * DIMM;
        cvt_f32_bf16<<<(unsigned)((n + 255) / 256), 256, 0, stream>>>(embed, embed_bf, n);
        n = (size_t)DEPTHL * DPROJP * DIMM;
        cvt_w_in_pad<<<(unsigned)((n + 255) / 256), 256, 0, stream>>>(w_in, w_in_bf);
        n = (size_t)DEPTHL * DIMM * DINN;
        cvt_f32_bf16<<<(unsigned)((n + 255) / 256), 256, 0, stream>>>(w_out, w_out_bf, n);
    }

    // h = embed[tokens]
    embed_gather<<<ROWSM, 256, 0, stream>>>(tokens, embed, h_f, h_bf);

    for (int l = 0; l < DEPTHL; ++l) {
        // in-proj: zxbcdt[2048,2192] = h_bf x w_in_l^T   (W rows padded to 2304)
        {
            dim3 grid(ROWSM / BM, DPROJP / BN);   // 64 x 9
            gemm_bf16_wmma<<<grid, 256, 0, stream>>>(
                h_bf, w_in_bf + (size_t)l * DPROJP * DIMM,
                zxbcdt, nullptr, nullptr, ROWSM, DPROJJ, DIMM);
        }
        // depthwise conv + SiLU
        conv_silu<<<(ROWSM * CONVD + 255) / 256, 256, 0, stream>>>(
            zxbcdt, conv_w + (size_t)l * CONVD * DCONVK, conv_b + (size_t)l * CONVD, xbc);
        // selective-state scan (+ D*x skip)
        ssm_scan<<<BATCHB * NHEAD, 256, 0, stream>>>(
            zxbcdt, xbc, dt_bias + l * NHEAD, A_log + l * NHEAD, D_skip + l * NHEAD, ybuf);
        // gated RMSNorm -> bf16
        gated_rmsnorm<<<ROWSM, 256, 0, stream>>>(zxbcdt, ybuf, norm_w + (size_t)l * DINN, yg_bf);
        // out-proj + residual: h_f += yg_bf x w_out_l^T ; refresh h_bf
        {
            dim3 grid(ROWSM / BM, DIMM / BN);     // 64 x 2
            gemm_bf16_wmma<<<grid, 256, 0, stream>>>(
                yg_bf, w_out_bf + (size_t)l * DIMM * DINN,
                h_f, h_bf, h_f, ROWSM, DIMM, DINN);
        }
    }

    // final norm -> bf16
    final_rmsnorm<<<ROWSM, 256, 0, stream>>>(h_f, norm_f, hn_bf);

    // tied LM head: logits[2048,32000] = hn_bf x embed_bf^T
    {
        dim3 grid(ROWSM / BM, VOCAB / BN);        // 64 x 125
        gemm_bf16_wmma<<<grid, 256, 0, stream>>>(
            hn_bf, embed_bf, logits, nullptr, nullptr, ROWSM, VOCAB, DIMM);
    }
}
